// beit_unet_64682207477779
// MI455X (gfx1250) — compile-verified, matrix operand order checked by reading, not run on hardware
//
#include <hip/hip_runtime.h>
#include <hip/hip_bf16.h>
#include <math.h>

#define SDIM   197
#define SPAD   224
#define DMODEL 768
#define DFF    3072
#define NHEAD  12
#define HDIM   64
#define NLAYER 3
#define NDREL  732
#define BATCH  64
#define NTOK   (BATCH * SDIM)   // 12608 = 197 * 64

typedef _Float16 half8  __attribute__((ext_vector_type(8)));
typedef _Float16 half16 __attribute__((ext_vector_type(16)));
typedef float    float8 __attribute__((ext_vector_type(8)));

// Load one 16x32 (or 32x16 transposed) f16 WMMA fragment from LDS.
// Per CDNA5 ISA: lanes 0-15 hold K[0..7] (vgpr0-3) and K[16..23] (vgpr4-7);
// lanes 16-31 hold K[8..15] and K[24..31]. rowp points at this lane's row.
__device__ __forceinline__ half16 lds_frag(const _Float16* rowp, int lane) {
  const int khalf = (lane & 16) ? 8 : 0;
  half8 lo = *(const half8*)(rowp + khalf);
  half8 hi = *(const half8*)(rowp + 16 + khalf);
  half16 r;
#pragma unroll
  for (int j = 0; j < 8; ++j) { r[j] = lo[j]; r[j + 8] = hi[j]; }
  return r;
}

__device__ __forceinline__ float geluf(float x) {
  return 0.5f * x * (1.0f + erff(x * 0.70710678118654752f));
}

// BEiT relative-position index (host constant reproduced on device)
__device__ __forceinline__ int rel_index(int q, int k) {
  if (q == 0 && k == 0) return NDREL - 1;   // 731
  if (k == 0) return NDREL - 2;             // 730
  if (q == 0) return NDREL - 3;             // 729
  int qi = (q - 1) / 14, qj = (q - 1) % 14;
  int ki = (k - 1) / 14, kj = (k - 1) % 14;
  return (qi - ki + 13) * 27 + (qj - kj + 13);
}

// ---------------- patch embed: identity reshape + cls prepend ---------------
__global__ __launch_bounds__(256)
void k_embed(const float* __restrict__ x, const float* __restrict__ cls,
             float* __restrict__ h) {
  size_t i = (size_t)blockIdx.x * 256 + threadIdx.x;
  if (i >= (size_t)NTOK * DMODEL) return;
  int c = (int)(i % DMODEL);
  size_t t = i / DMODEL;
  int s = (int)(t % SDIM);
  size_t b = t / SDIM;
  h[i] = (s == 0) ? cls[c] : x[(b * 196 + (size_t)(s - 1)) * DMODEL + c];
}

// ------ expand rel_tbl[layer][REL_IDX] into dense [H][SPAD][SPAD] table -----
// Does the integer divisions exactly once per (q,k) instead of per score tile.
__global__ __launch_bounds__(256)
void k_relbias(const float* __restrict__ rel, float* __restrict__ relb,
               int layer) {
  int i = blockIdx.x * 256 + threadIdx.x;          // over SDIM*SDIM
  if (i >= SDIM * SDIM) return;
  const int q = i / SDIM, k = i % SDIM;
  const int idx = rel_index(q, k);
  const float* src = rel + ((size_t)layer * NDREL + idx) * NHEAD;
  float* dst = relb + (size_t)q * SPAD + k;
#pragma unroll
  for (int h = 0; h < NHEAD; ++h)
    dst[(size_t)h * SPAD * SPAD] = src[h];
}

// ---------------- layernorm: one wave32 per 768-wide row --------------------
__global__ __launch_bounds__(256)
void k_ln(const float* __restrict__ x, const float* __restrict__ g,
          const float* __restrict__ bb, float* __restrict__ y, int rows) {
  const int lane = threadIdx.x & 31;
  const int row  = blockIdx.x * 8 + (threadIdx.x >> 5);
  if (row >= rows) return;
  const float* xr = x + (size_t)row * DMODEL;
  float vals[24];
  float s = 0.f;
#pragma unroll
  for (int j = 0; j < 24; ++j) { vals[j] = xr[lane + 32 * j]; s += vals[j]; }
#pragma unroll
  for (int o = 16; o > 0; o >>= 1) s += __shfl_xor(s, o, 32);
  const float mean = s * (1.0f / DMODEL);
  float vs = 0.f;
#pragma unroll
  for (int j = 0; j < 24; ++j) { float d = vals[j] - mean; vs += d * d; }
#pragma unroll
  for (int o = 16; o > 0; o >>= 1) vs += __shfl_xor(vs, o, 32);
  const float inv = rsqrtf(vs * (1.0f / DMODEL) + 1e-12f);
  float* yr = y + (size_t)row * DMODEL;
#pragma unroll
  for (int j = 0; j < 24; ++j) {
    int c = lane + 32 * j;
    yr[c] = (vals[j] - mean) * inv * g[c] + bb[c];
  }
}

// ---------------- main WMMA GEMM: C[M,N] = act(A[M,K] @ W[K,N] + bias) ------
// optional residual: C = res + C * lam (per-column lambda)
// N, K compile-time (immediate-offset loads, constant strides).
// 256 threads = 8 waves. Block tile 64(M) x 128(N); each wave 16(M) x 64(N).
// Double-buffered LDS: one barrier per K-step, next tile's global loads are
// issued before the current tile's 4 WMMAs.
template <int N, int K, int ACT>
__global__ __launch_bounds__(256)
void k_gemm(const float* __restrict__ A, const float* __restrict__ W,
            const float* __restrict__ bias, const float* __restrict__ res,
            const float* __restrict__ lam, float* __restrict__ C) {
  constexpr int T = K / 32;
  __shared__ __attribute__((aligned(16))) _Float16 As[2][64][32];    // [m][k]
  __shared__ __attribute__((aligned(16))) _Float16 Bs[2][128][32];   // [n][k]
  const int tid  = threadIdx.x;
  const int lane = tid & 31;
  const int wid  = tid >> 5;
  const int m_base = (wid & 3) * 16;    // 4 waves along M
  const int n_base = (wid >> 2) * 64;   // 2 waves along N, 64 cols each
  const int rowM = blockIdx.y * 64;
  const int colN = blockIdx.x * 128;

  const int ar = tid >> 2;              // 0..63
  const int ac = (tid & 3) * 8;         // 0,8,16,24
  const int bn  = tid & 127;            // 0..127 : column of W this thread owns
  const int bkg = (tid >> 7) * 16;      // 0 or 16 : k-subrange

  const float* aptr = A + (size_t)(rowM + ar) * K + ac;
  const float* wptr = W + (size_t)bkg * N + colN + bn;

  auto stage = [&](int buf, const float* ap_, const float* wp_) {
    // A tile: two b128 loads -> packed cvt -> one b128 LDS store
    float4 a0 = ((const float4*)ap_)[0];
    float4 a1 = ((const float4*)ap_)[1];
    half8 ah;
    ah[0] = (_Float16)a0.x; ah[1] = (_Float16)a0.y;
    ah[2] = (_Float16)a0.z; ah[3] = (_Float16)a0.w;
    ah[4] = (_Float16)a1.x; ah[5] = (_Float16)a1.y;
    ah[6] = (_Float16)a1.z; ah[7] = (_Float16)a1.w;
    *(half8*)&As[buf][ar][ac] = ah;
    // B tile (n-major): 16 immediate-offset column loads from one base
    float wv[16];
#pragma unroll
    for (int j = 0; j < 16; ++j) wv[j] = wp_[j * N];
    half8 h0, h1;
#pragma unroll
    for (int j = 0; j < 8; ++j) {
      h0[j] = (_Float16)wv[j];
      h1[j] = (_Float16)wv[8 + j];
    }
    *(half8*)&Bs[buf][bn][bkg]     = h0;
    *(half8*)&Bs[buf][bn][bkg + 8] = h1;
  };

  float8 acc0 = {}, acc1 = {}, acc2 = {}, acc3 = {};

  stage(0, aptr, wptr);

  for (int kt = 0; kt < T; ++kt) {
    __syncthreads();                       // buf[kt&1] now visible; buf[!cur] free
    const int cur = kt & 1;
    if (kt + 1 < T)
      stage(cur ^ 1, aptr + (size_t)(kt + 1) * 32,
            wptr + (size_t)(kt + 1) * 32 * N);
    if (kt + 2 < T) {                      // global_prefetch_b8, two tiles ahead
      __builtin_prefetch(aptr + (size_t)(kt + 2) * 32, 0, 3);
      __builtin_prefetch(wptr + (size_t)(kt + 2) * 32 * N, 0, 3);
    }

    half16 af  = lds_frag(&As[cur][m_base + (lane & 15)][0], lane);
    half16 bf0 = lds_frag(&Bs[cur][n_base +  0 + (lane & 15)][0], lane);
    half16 bf1 = lds_frag(&Bs[cur][n_base + 16 + (lane & 15)][0], lane);
    half16 bf2 = lds_frag(&Bs[cur][n_base + 32 + (lane & 15)][0], lane);
    half16 bf3 = lds_frag(&Bs[cur][n_base + 48 + (lane & 15)][0], lane);
    acc0 = __builtin_amdgcn_wmma_f32_16x16x32_f16(false, af, false, bf0,
                                                  (short)0, acc0, false, false);
    acc1 = __builtin_amdgcn_wmma_f32_16x16x32_f16(false, af, false, bf1,
                                                  (short)0, acc1, false, false);
    acc2 = __builtin_amdgcn_wmma_f32_16x16x32_f16(false, af, false, bf2,
                                                  (short)0, acc2, false, false);
    acc3 = __builtin_amdgcn_wmma_f32_16x16x32_f16(false, af, false, bf3,
                                                  (short)0, acc3, false, false);
  }

  // epilogue: C/D layout — vgpr r: lanes 0-15 -> M=r, lanes 16-31 -> M=8+r
  const int mofs = m_base + ((lane & 16) ? 8 : 0);
  const int nl   = lane & 15;
  float8 accs[4] = {acc0, acc1, acc2, acc3};
#pragma unroll
  for (int t = 0; t < 4; ++t) {
    const int gn = colN + n_base + t * 16 + nl;
#pragma unroll
    for (int r = 0; r < 8; ++r) {
      const int gm = rowM + mofs + r;
      float v = accs[t][r];
      if (bias) v += bias[gn];
      if (ACT)  v = geluf(v);
      if (res)  v = res[(size_t)gm * N + gn] + v * lam[gn];
      C[(size_t)gm * N + gn] = v;
    }
  }
}

// ---------------- attention scores: q @ k^T * scale + relpos ---------------
// one block (128 thr = 4 waves) computes a 32x32 score tile for one (b,h).
// sc and relb are [.][SPAD][SPAD] row-padded: epilogue is fully unguarded
// (pad rows/cols are written with finite garbage and never read).
__global__ __launch_bounds__(128)
void k_scores(const float* __restrict__ q, const float* __restrict__ kk,
              const float* __restrict__ relb, float* __restrict__ sc) {
  __shared__ __attribute__((aligned(16))) _Float16 Qs[32][64];
  __shared__ __attribute__((aligned(16))) _Float16 Ks[32][64];
  const int tid = threadIdx.x, lane = tid & 31, wid = tid >> 5;
  const int bh = blockIdx.z;
  const int b = bh / NHEAD, h = bh % NHEAD;
  const int qt = blockIdx.y * 32, kt = blockIdx.x * 32;

  const int lr = tid >> 2;          // 0..31
  const int lc = (tid & 3) * 16;    // 0,16,32,48
  {
    const int qrow = qt + lr;
    const float* sq = q + ((size_t)(b * SDIM + (qrow < SDIM ? qrow : 0)) * DMODEL
                           + h * HDIM + lc);
#pragma unroll
    for (int j = 0; j < 16; ++j)
      Qs[lr][lc + j] = (qrow < SDIM) ? (_Float16)sq[j] : (_Float16)0.f;
    const int krow = kt + lr;
    const float* sk = kk + ((size_t)(b * SDIM + (krow < SDIM ? krow : 0)) * DMODEL
                            + h * HDIM + lc);
#pragma unroll
    for (int j = 0; j < 16; ++j)
      Ks[lr][lc + j] = (krow < SDIM) ? (_Float16)sk[j] : (_Float16)0.f;
  }
  __syncthreads();

  const int m_base = (wid & 1) * 16;
  const int n_base = (wid >> 1) * 16;
  float8 acc = {};
#pragma unroll
  for (int kc = 0; kc < 2; ++kc) {
    half16 af = lds_frag(&Qs[m_base + (lane & 15)][kc * 32], lane);
    half16 bf = lds_frag(&Ks[n_base + (lane & 15)][kc * 32], lane);
    acc = __builtin_amdgcn_wmma_f32_16x16x32_f16(false, af, false, bf,
                                                 (short)0, acc, false, false);
  }

  const float scale = 0.125f;  // 1/sqrt(64)
  const int mofs = m_base + ((lane & 16) ? 8 : 0);
  const int n = n_base + (lane & 15);
  // base offset for r = 0; consecutive r -> +SPAD (compile-time stride)
  const size_t o0 = (size_t)(qt + mofs) * SPAD + (kt + n);
  const float* rb = relb + (size_t)h * SPAD * SPAD + o0;
  float* sd = sc + (size_t)bh * SPAD * SPAD + o0;
  float bvals[8];
#pragma unroll
  for (int r = 0; r < 8; ++r) bvals[r] = rb[r * SPAD];   // clause-batched
#pragma unroll
  for (int r = 0; r < 8; ++r) sd[r * SPAD] = acc[r] * scale + bvals[r];
}

// ---------------- softmax over 197 cols (224-padded), wave per row ----------
// grid: (ceil(SDIM/8), BATCH*NHEAD); sc rows are SPAD-padded per (b,h).
__global__ __launch_bounds__(256)
void k_softmax(float* __restrict__ sc) {
  const int lane = threadIdx.x & 31;
  const int q    = blockIdx.x * 8 + (threadIdx.x >> 5);
  if (q >= SDIM) return;
  const int bh = blockIdx.y;
  float* r = sc + ((size_t)bh * SPAD + q) * SPAD;
  float v[7];
  float mx = -1e30f;
#pragma unroll
  for (int j = 0; j < 7; ++j) {
    const int c = lane + 32 * j;
    v[j] = (c < SDIM) ? r[c] : -1e30f;
    mx = fmaxf(mx, v[j]);
  }
#pragma unroll
  for (int o = 16; o > 0; o >>= 1) mx = fmaxf(mx, __shfl_xor(mx, o, 32));
  float s = 0.f;
#pragma unroll
  for (int j = 0; j < 7; ++j) {
    const int c = lane + 32 * j;
    v[j] = (c < SDIM) ? __expf(v[j] - mx) : 0.f;
    s += v[j];
  }
#pragma unroll
  for (int o = 16; o > 0; o >>= 1) s += __shfl_xor(s, o, 32);
  const float inv = 1.f / s;
#pragma unroll
  for (int j = 0; j < 7; ++j) r[lane + 32 * j] = v[j] * inv;  // pads get 0
}

// ---------------- context: probs[197,224] @ v_head[224,64] ------------------
// block (128 thr = 4 waves) computes 32(q) x 64(d) for one (b,h).
// Double-buffered LDS, one barrier per K-step. probs are [SPAD][SPAD]-padded.
__global__ __launch_bounds__(128)
void k_ctx(const float* __restrict__ p, const float* __restrict__ v,
           float* __restrict__ ctx) {
  constexpr int T = SPAD / 32;  // 7
  __shared__ __attribute__((aligned(16))) _Float16 Ps[2][32][32];  // [q][k]
  __shared__ __attribute__((aligned(16))) _Float16 Vs[2][64][32];  // [d][k]
  const int tid = threadIdx.x, lane = tid & 31, wid = tid >> 5;
  const int bh = blockIdx.y;
  const int b = bh / NHEAD, h = bh % NHEAD;
  const int qt = blockIdx.x * 32;
  const int m_base = (wid & 1) * 16;
  const int n_base = (wid >> 1) * 32;

  const int pr = tid >> 2, pc = (tid & 3) * 8;    // probs staging
  const int vk = tid >> 2, vd = (tid & 3) * 16;   // v staging (transposed)

  auto stage = [&](int buf, int k0) {
    const int qrow = qt + pr;
    const float* sp = p + ((size_t)bh * SPAD + (qrow < SDIM ? qrow : 0)) * SPAD
                        + k0 + pc;
#pragma unroll
    for (int j = 0; j < 8; ++j)
      Ps[buf][pr][pc + j] = (qrow < SDIM) ? (_Float16)sp[j] : (_Float16)0.f;
    const int krow = k0 + vk;
    const float* sv = v + ((size_t)(b * SDIM + (krow < SDIM ? krow : 0)) * DMODEL
                           + h * HDIM + vd);
#pragma unroll
    for (int j = 0; j < 16; ++j)
      Vs[buf][vd + j][vk] = (krow < SDIM) ? (_Float16)sv[j] : (_Float16)0.f;
  };

  float8 acc0 = {};
  float8 acc1 = {};

  stage(0, 0);

  for (int kt = 0; kt < T; ++kt) {
    __syncthreads();
    const int cur = kt & 1;
    if (kt + 1 < T) stage(cur ^ 1, (kt + 1) * 32);

    half16 af  = lds_frag(&Ps[cur][m_base + (lane & 15)][0], lane);
    half16 bf0 = lds_frag(&Vs[cur][n_base + (lane & 15)][0], lane);
    half16 bf1 = lds_frag(&Vs[cur][n_base + 16 + (lane & 15)][0], lane);
    acc0 = __builtin_amdgcn_wmma_f32_16x16x32_f16(false, af, false, bf0,
                                                  (short)0, acc0, false, false);
    acc1 = __builtin_amdgcn_wmma_f32_16x16x32_f16(false, af, false, bf1,
                                                  (short)0, acc1, false, false);
  }

  const int mofs = m_base + ((lane & 16) ? 8 : 0);
  const int d0 = n_base + (lane & 15);
#pragma unroll
  for (int r = 0; r < 8; ++r) {
    const int gq = qt + mofs + r;
    if (gq < SDIM) {
      float* dst = ctx + (size_t)(b * SDIM + gq) * DMODEL + h * HDIM;
      dst[d0] = acc0[r];
      dst[d0 + 16] = acc1[r];
    }
  }
}

// ---------------------------------------------------------------------------
extern "C" void kernel_launch(void* const* d_in, const int* in_sizes, int n_in,
                              void* d_out, int out_size, void* d_ws,
                              size_t ws_size, hipStream_t stream) {
  const float* x    = (const float*)d_in[0];
  const float* cls  = (const float*)d_in[1];
  const float* g1   = (const float*)d_in[2];
  const float* b1   = (const float*)d_in[3];
  const float* Wq   = (const float*)d_in[4];
  const float* bq   = (const float*)d_in[5];
  const float* Wk   = (const float*)d_in[6];
  const float* Wv   = (const float*)d_in[7];
  const float* bv   = (const float*)d_in[8];
  const float* rel  = (const float*)d_in[9];
  const float* Wo   = (const float*)d_in[10];
  const float* bo   = (const float*)d_in[11];
  const float* lam1 = (const float*)d_in[12];
  const float* g2   = (const float*)d_in[13];
  const float* b2   = (const float*)d_in[14];
  const float* Wi   = (const float*)d_in[15];
  const float* bi   = (const float*)d_in[16];
  const float* Wmo  = (const float*)d_in[17];
  const float* bmo  = (const float*)d_in[18];
  const float* lam2 = (const float*)d_in[19];
  float* out = (float*)d_out;

  float* ws = (float*)d_ws;
  const size_t nTD = (size_t)NTOK * DMODEL;
  float* h    = ws; ws += nTD;
  float* hn   = ws; ws += nTD;
  float* qb   = ws; ws += nTD;
  float* kb   = ws; ws += nTD;
  float* vb   = ws; ws += nTD;
  float* ctx  = ws; ws += nTD;
  float* sc   = ws; ws += (size_t)BATCH * NHEAD * SPAD * SPAD;
  float* ff   = ws; ws += (size_t)NTOK * DFF;
  float* relb = ws; ws += (size_t)NHEAD * SPAD * SPAD;

  const dim3 b256(256);
  k_embed<<<dim3((unsigned)((nTD + 255) / 256)), b256, 0, stream>>>(x, cls, h);

  for (int i = 0; i < NLAYER; ++i) {
    const size_t oM  = (size_t)i * DMODEL * DMODEL;
    const size_t oV  = (size_t)i * DMODEL;
    const size_t oWi = (size_t)i * DMODEL * DFF;
    const size_t oBi = (size_t)i * DFF;

    // --- attention block ---
    k_ln<<<dim3(NTOK / 8), b256, 0, stream>>>(h, g1 + oV, b1 + oV, hn, NTOK);
    k_relbias<<<dim3((SDIM * SDIM + 255) / 256), b256, 0, stream>>>(rel, relb,
                                                                    i);

    const dim3 gQ(DMODEL / 128, NTOK / 64);   // (6, 197)
    k_gemm<DMODEL, DMODEL, 0><<<gQ, b256, 0, stream>>>(
        hn, Wq + oM, bq + oV, nullptr, nullptr, qb);
    k_gemm<DMODEL, DMODEL, 0><<<gQ, b256, 0, stream>>>(
        hn, Wk + oM, nullptr, nullptr, nullptr, kb);
    k_gemm<DMODEL, DMODEL, 0><<<gQ, b256, 0, stream>>>(
        hn, Wv + oM, bv + oV, nullptr, nullptr, vb);

    k_scores<<<dim3(7, 7, BATCH * NHEAD), dim3(128), 0, stream>>>(qb, kb, relb,
                                                                  sc);
    k_softmax<<<dim3((SDIM + 7) / 8, BATCH * NHEAD), b256, 0, stream>>>(sc);
    k_ctx<<<dim3(7, BATCH * NHEAD), dim3(128), 0, stream>>>(sc, vb, ctx);

    // out-proj with fused residual: h = h + (ctx@Wo + bo) * lam1
    k_gemm<DMODEL, DMODEL, 0><<<gQ, b256, 0, stream>>>(
        ctx, Wo + oM, bo + oV, h, lam1 + oV, h);

    // --- MLP block ---
    k_ln<<<dim3(NTOK / 8), b256, 0, stream>>>(h, g2 + oV, b2 + oV, hn, NTOK);

    const dim3 gF(DFF / 128, NTOK / 64);      // (24, 197)
    k_gemm<DFF, DMODEL, 1><<<gF, b256, 0, stream>>>(
        hn, Wi + oWi, bi + oBi, nullptr, nullptr, ff);  // GELU fused

    float* dst = (i == NLAYER - 1) ? out : h;
    k_gemm<DMODEL, DFF, 0><<<gQ, b256, 0, stream>>>(
        ff, Wmo + oWi, bmo + oV, h, lam2 + oV, dst);
  }
}
